// SelfLayer_36112085025170
// MI455X (gfx1250) — matrix operand order repaired and checked by reading, not compiled, not run on hardware
//
#include <hip/hip_runtime.h>
#include <math.h>
#include <stdint.h>

// ---------------------------------------------------------------------------
// QHNet SelfLayer, fused for MI455X (gfx1250, wave32, WMMA).
//
// Roofline: ~196 GFLOP total, ~307 MB mandatory HBM traffic (13 us @ 23.3TB/s).
// Weights (~3.5 MB bf16) are shared by all 50000 nodes -> L2 resident.
// All matmuls run as V_WMMA_F32_16X16X32_BF16 (bf16 in, f32 accumulate).
// Each block = 32 nodes (8 wave32); each wave holds two row-half accumulators
// so every B (weight) fragment fetched from L2 feeds 2 WMMAs.
// All global accesses go through explicit addrspace(1) pointers so the
// backend emits global_load/global_store (not flat_*, which also burns DScnt).
// ---------------------------------------------------------------------------

typedef __bf16 bf16_t;
typedef __attribute__((ext_vector_type(16))) __bf16 v16bf;
typedef __attribute__((ext_vector_type(8)))  __bf16 v8bf;
typedef __attribute__((ext_vector_type(8)))  float  v8f;

// explicit global-address-space views (built via inttoptr -> always legal)
typedef __attribute__((address_space(1))) const v16bf  gc_v16bf;
typedef __attribute__((address_space(1))) const float  gc_float;
typedef __attribute__((address_space(1))) const bf16_t gc_bf16;
typedef __attribute__((address_space(1)))       float  g_float;

__device__ __forceinline__ gc_float* gfp(const float* p) {
  return (gc_float*)(uintptr_t)p;
}
__device__ __forceinline__ gc_bf16* gbp(const bf16_t* p) {
  return (gc_bf16*)(uintptr_t)p;
}
__device__ __forceinline__ g_float* gfw(float* p) {
  return (g_float*)(uintptr_t)p;
}

// global_prefetch_b8: no result, no counter -> safe as raw asm, and it cannot
// block address-space inference the way llvm.prefetch did.
__device__ __forceinline__ void prefetch_g(const void* p) {
  asm volatile("global_prefetch_b8 %0, off scope:SCOPE_SE"
               :: "v"((unsigned long long)(uintptr_t)p));
}

#define TILE_M   32
#define NTHREADS 256
#define NWAVES   8

__device__ __forceinline__ float silu_f(float x) { return x / (1.0f + __expf(-x)); }

__device__ __forceinline__ v8f wmma_bf16(v16bf a, v16bf b, v8f c) {
  // (neg_a, A, neg_b, B, c_mod, C, reuse_a, reuse_b)
  return __builtin_amdgcn_wmma_f32_16x16x32_bf16(false, a, false, b, (short)0, c,
                                                 false, false);
}

// A-fragment (16x32 bf16) from LDS, row-major [16][lda].
// Contiguous k order, as in glb_load_b: lanes 0-15 hold row M=lane, K = k0+0..15;
//             lanes 16-31 hold row M=lane-16, K = k0+16..31.
__device__ __forceinline__ v16bf lds_load_a(const bf16_t* base, int lda, int k0,
                                            int lane) {
  const int row = lane & 15;
  const int kb  = k0 + (lane & 16);             // +16 halves on upper half-wave
  const bf16_t* p = base + row * lda + kb;
  union { v16bf v; v8bf h[2]; } u;
  u.h[0] = *(const v8bf*)(p);
  u.h[1] = *(const v8bf*)(p + 8);
  return u.v;
}

// B-fragment (32x16 bf16) from global, weights stored transposed Wt[n][k].
// lanes 0-15: column N=lane, K = k0..k0+15; lanes 16-31: K = k0+16..k0+31.
// -> one contiguous 32B load per lane through an addrspace(1) pointer.
__device__ __forceinline__ v16bf glb_load_b(const bf16_t* wrow, int k0) {
  return *(gc_v16bf*)(uintptr_t)(wrow + k0);
}

// One 512->512 MLP layer on a 32-row tile held in LDS (two 16-row halves).
// MODE 0: dstH[n][col] = bf16(silu(acc + bias[col]))            (first FC)
// MODE 1: NormGate output: col<256 -> gs = silu(g);
//         col>=256 -> gv[i] = vsrc[i] * g (per vector component) (second FC)
template <int MODE>
__device__ void mlp_layer(const bf16_t* __restrict__ A,
                          const bf16_t* __restrict__ Wt,
                          const float*  __restrict__ bias,
                          bf16_t* __restrict__ dstH,
                          bf16_t* __restrict__ gs,
                          bf16_t* __restrict__ gv,
                          const bf16_t* __restrict__ vsrc,
                          int wave, int lane) {
  const int colsub = lane & 15;
  const int rbase  = (lane & 16) >> 1;
  for (int t = wave; t < 32; t += NWAVES) {
    const int n0 = t * 16;
    const bf16_t* wrow = Wt + (size_t)(n0 + colsub) * 512 + (lane & 16);
    v8f acc[2] = {};
#pragma unroll 4
    for (int k0 = 0; k0 < 512; k0 += 32) {
      if (k0 + 32 < 512) prefetch_g(wrow + k0 + 32);
      v16bf b  = glb_load_b(wrow, k0);             // 1 weight fetch ...
      v16bf a0 = lds_load_a(A, 512, k0, lane);
      v16bf a1 = lds_load_a(A + 16 * 512, 512, k0, lane);
      acc[0] = wmma_bf16(a0, b, acc[0]);           // ... feeds 2 WMMAs
      acc[1] = wmma_bf16(a1, b, acc[1]);
    }
    const int   colg = n0 + colsub;
    const float bv   = gfp(bias)[colg];
#pragma unroll
    for (int h = 0; h < 2; ++h) {
      if (MODE == 0) {
#pragma unroll
        for (int j = 0; j < 8; ++j) {
          float g = acc[h][j] + bv;
          dstH[(h * 16 + rbase + j) * 512 + colg] = (bf16_t)silu_f(g);
        }
      } else {
        if (colg < 256) {
#pragma unroll
          for (int j = 0; j < 8; ++j) {
            float g = acc[h][j] + bv;
            gs[(h * 16 + rbase + j) * 256 + colg] = (bf16_t)silu_f(g);
          }
        } else {
          const int ch = colg - 256;
#pragma unroll
          for (int j = 0; j < 8; ++j) {
            float g = acc[h][j] + bv;
#pragma unroll
            for (int i = 0; i < 3; ++i) {
              const int o = i * (32 * 256) + (h * 16 + rbase + j) * 256 + ch;
              gv[o] = (bf16_t)((float)vsrc[o] * g);
            }
          }
        }
      }
    }
  }
}

// o3.Linear core: scalar path + 3 vector components x 2 row-halves for one
// 16-column tile. K = 256. W0t/W1t stored transposed [128][256] bf16.
__device__ __forceinline__ void o3_accum(const bf16_t* __restrict__ gs,
                                         const bf16_t* __restrict__ gv,
                                         const bf16_t* __restrict__ W0t,
                                         const bf16_t* __restrict__ W1t,
                                         int n0, int lane,
                                         v8f accS[2], v8f accV[3][2]) {
  const int colsub = lane & 15;
  const bf16_t* w0row = W0t + (size_t)(n0 + colsub) * 256 + (lane & 16);
  const bf16_t* w1row = W1t + (size_t)(n0 + colsub) * 256 + (lane & 16);
#pragma unroll 2
  for (int k0 = 0; k0 < 256; k0 += 32) {
    if (k0 + 32 < 256) {
      prefetch_g(w0row + k0 + 32);
      prefetch_g(w1row + k0 + 32);
    }
    v16bf bw0 = glb_load_b(w0row, k0);
    accS[0] = wmma_bf16(lds_load_a(gs, 256, k0, lane), bw0, accS[0]);
    accS[1] = wmma_bf16(lds_load_a(gs + 16 * 256, 256, k0, lane), bw0, accS[1]);
    v16bf bw1 = glb_load_b(w1row, k0);
#pragma unroll
    for (int i = 0; i < 3; ++i) {
      const bf16_t* gvi = gv + i * (32 * 256);
      accV[i][0] = wmma_bf16(lds_load_a(gvi, 256, k0, lane), bw1, accV[i][0]);
      accV[i][1] =
          wmma_bf16(lds_load_a(gvi + 16 * 256, 256, k0, lane), bw1, accV[i][1]);
    }
  }
}

// ---------------------------------------------------------------------------
// Prep: W[k][n] (f32 row-major) -> Wt[n][k] (bf16 row-major)
// ---------------------------------------------------------------------------
__global__ void __launch_bounds__(256)
transpose_bf16_kernel(const float* __restrict__ W, bf16_t* __restrict__ Wt,
                      int K, int N) {
  int idx = blockIdx.x * blockDim.x + threadIdx.x;
  if (idx >= K * N) return;
  int n = idx / K;
  int k = idx - n * K;
  ((__attribute__((address_space(1))) bf16_t*)(uintptr_t)Wt)[idx] =
      (bf16_t)gfp(W)[(size_t)k * N + n];
}

// ---------------------------------------------------------------------------
// Fused SelfLayer kernel. One block (8 wave32) = 32 nodes, whole layer in LDS.
// ---------------------------------------------------------------------------
__global__ void __launch_bounds__(NTHREADS)
selflayer_kernel(const float* __restrict__ x,
                 const float* __restrict__ fcl1_b, const float* __restrict__ fcl2_b,
                 const float* __restrict__ fcr1_b, const float* __restrict__ fcr2_b,
                 const float* __restrict__ fcp1_b, const float* __restrict__ fcp2_b,
                 const float* __restrict__ bl0, const float* __restrict__ br0,
                 const float* __restrict__ bp0,
                 const float* __restrict__ w_ss, const float* __restrict__ w_sv,
                 const float* __restrict__ w_vs, const float* __restrict__ w_vv,
                 const bf16_t* __restrict__ fcl1_t, const bf16_t* __restrict__ fcl2_t,
                 const bf16_t* __restrict__ fcr1_t, const bf16_t* __restrict__ fcr2_t,
                 const bf16_t* __restrict__ fcp1_t, const bf16_t* __restrict__ fcp2_t,
                 const bf16_t* __restrict__ Wl0_t, const bf16_t* __restrict__ Wl1_t,
                 const bf16_t* __restrict__ Wr0_t, const bf16_t* __restrict__ Wr1_t,
                 const bf16_t* __restrict__ Wp0_t, const bf16_t* __restrict__ Wp1_t,
                 float* __restrict__ out, int nrows) {
  extern __shared__ char smem_raw[];
  bf16_t* s0   = (bf16_t*)smem_raw;      // [32][256]     tp_s overwrites later
  bf16_t* v0   = s0   + 32 * 256;        // [3][32][256]  tp_v overwrites later
  bf16_t* abuf = v0   + 3 * 32 * 256;    // [32][512]     concat(s, |v|) bf16
  bf16_t* hbuf = abuf + 32 * 512;        // [32][512]     silu(fc1 out)
  bf16_t* gs   = hbuf + 32 * 512;        // [32][256]     gated scalars
  bf16_t* gv   = gs   + 32 * 256;        // [3][32][256]  gated vectors
  bf16_t* LSb  = gv   + 3 * 32 * 256;    // [32][128]     left o3 scalar out
  bf16_t* LVb  = LSb  + 32 * 128;        // [3][32][128]  left o3 vector out

  const int tid  = threadIdx.x;
  const int wave = tid >> 5;
  const int lane = tid & 31;
  const int row0 = blockIdx.x * TILE_M;
  const float inv       = 0.0625f;          // 1/sqrt(256) fan-in norm
  const float inv_sqrt3 = 0.57735026919f;   // CG norm 1o x 1o -> 0e

  // ---- Stage 0: load x tile; build s0, v0 (component-major), abuf -----------
  for (int idx = tid; idx < 32 * 256; idx += NTHREADS) {
    const int n = idx >> 8, c = idx & 255;
    int r = row0 + n;
    if (r >= nrows) r = nrows - 1;
    gc_float* xr = gfp(x) + (size_t)r * 1024;
    const float sval = xr[c];
    s0[n * 256 + c]   = (bf16_t)sval;
    abuf[n * 512 + c] = (bf16_t)sval;
    const float a  = xr[256 + 3 * c + 0];
    const float b  = xr[256 + 3 * c + 1];
    const float cc = xr[256 + 3 * c + 2];
    v0[0 * 8192 + n * 256 + c] = (bf16_t)a;
    v0[1 * 8192 + n * 256 + c] = (bf16_t)b;
    v0[2 * 8192 + n * 256 + c] = (bf16_t)cc;
    abuf[n * 512 + 256 + c] = (bf16_t)sqrtf(a * a + b * b + cc * cc);
  }
  __syncthreads();

  // ---- LEFT branch: NormGate MLP + o3.Linear --------------------------------
  mlp_layer<0>(abuf, fcl1_t, fcl1_b, hbuf, nullptr, nullptr, nullptr, wave, lane);
  __syncthreads();
  mlp_layer<1>(hbuf, fcl2_t, fcl2_b, nullptr, gs, gv, v0, wave, lane);
  __syncthreads();
  {
    const int n0 = wave * 16;  // 8 waves cover 8 column tiles of 128
    v8f accS[2] = {};
    v8f accV[3][2] = {};
    o3_accum(gs, gv, Wl0_t, Wl1_t, n0, lane, accS, accV);
    const int colh  = n0 + (lane & 15);
    const int rbase = (lane & 16) >> 1;
    const float b0v = gfp(bl0)[colh];
#pragma unroll
    for (int h = 0; h < 2; ++h)
#pragma unroll
      for (int j = 0; j < 8; ++j) {
        const int n = h * 16 + rbase + j;
        LSb[n * 128 + colh] = (bf16_t)(accS[h][j] * inv + b0v);
#pragma unroll
        for (int i = 0; i < 3; ++i)
          LVb[i * 4096 + n * 128 + colh] = (bf16_t)(accV[i][h][j] * inv);
      }
  }
  __syncthreads();

  // ---- RIGHT branch (same abuf input) + fused CG tensor product + residual --
  mlp_layer<0>(abuf, fcr1_t, fcr1_b, hbuf, nullptr, nullptr, nullptr, wave, lane);
  __syncthreads();
  mlp_layer<1>(hbuf, fcr2_t, fcr2_b, nullptr, gs, gv, v0, wave, lane);
  __syncthreads();
  {
    const int n0 = wave * 16;
    v8f accS[2] = {};
    v8f accV[3][2] = {};
    o3_accum(gs, gv, Wr0_t, Wr1_t, n0, lane, accS, accV);
    const int colh  = n0 + (lane & 15);
    const int rbase = (lane & 16) >> 1;
    const float b0v = gfp(br0)[colh];
    const float wss = gfp(w_ss)[colh], wsv = gfp(w_sv)[colh];
    const float wvs = gfp(w_vs)[colh], wvv = gfp(w_vv)[colh];
#pragma unroll
    for (int h = 0; h < 2; ++h)
#pragma unroll
      for (int j = 0; j < 8; ++j) {
        const int n = h * 16 + rbase + j;
        const float rs = accS[h][j] * inv + b0v;
        float rv[3], lv[3];
#pragma unroll
        for (int i = 0; i < 3; ++i) {
          rv[i] = accV[i][h][j] * inv;
          lv[i] = (float)LVb[i * 4096 + n * 128 + colh];
        }
        const float ls  = (float)LSb[n * 128 + colh];
        const float dot = lv[0] * rv[0] + lv[1] * rv[1] + lv[2] * rv[2];
        // tp_s = concat(w_ss*ls*rs, w_vv*(lv.rv)/sqrt3) + s   (overwrite s0)
        const float s_a = wss * ls * rs;
        const float s_b = wvv * dot * inv_sqrt3;
        const float os_a = (float)s0[n * 256 + colh];
        const float os_b = (float)s0[n * 256 + 128 + colh];
        s0[n * 256 + colh]       = (bf16_t)(s_a + os_a);
        s0[n * 256 + 128 + colh] = (bf16_t)(s_b + os_b);
        // tp_v = concat(w_sv*ls*rv, w_vs*lv*rs) + v           (overwrite v0)
#pragma unroll
        for (int i = 0; i < 3; ++i) {
          const float va = wsv * ls * rv[i];
          const float vb = wvs * lv[i] * rs;
          const float oa = (float)v0[i * 8192 + n * 256 + colh];
          const float ob = (float)v0[i * 8192 + n * 256 + 128 + colh];
          v0[i * 8192 + n * 256 + colh]       = (bf16_t)(va + oa);
          v0[i * 8192 + n * 256 + 128 + colh] = (bf16_t)(vb + ob);
        }
      }
  }
  __syncthreads();

  // ---- Stage 2: rebuild abuf = concat(tp_s, |tp_v|) -------------------------
  for (int idx = tid; idx < 32 * 256; idx += NTHREADS) {
    const int n = idx >> 8, c = idx & 255;
    abuf[n * 512 + c] = s0[n * 256 + c];
    const float a  = (float)v0[0 * 8192 + n * 256 + c];
    const float b  = (float)v0[1 * 8192 + n * 256 + c];
    const float cc = (float)v0[2 * 8192 + n * 256 + c];
    abuf[n * 512 + 256 + c] = (bf16_t)sqrtf(a * a + b * b + cc * cc);
  }
  __syncthreads();

  // ---- Output NormGate + projection, epilogue writes straight to global -----
  mlp_layer<0>(abuf, fcp1_t, fcp1_b, hbuf, nullptr, nullptr, nullptr, wave, lane);
  __syncthreads();
  mlp_layer<1>(hbuf, fcp2_t, fcp2_b, nullptr, gs, gv, v0, wave, lane);
  __syncthreads();
  {
    const int n0 = wave * 16;
    v8f accS[2] = {};
    v8f accV[3][2] = {};
    o3_accum(gs, gv, Wp0_t, Wp1_t, n0, lane, accS, accV);
    const int colh  = n0 + (lane & 15);
    const int rbase = (lane & 16) >> 1;
    const float b0v = gfp(bp0)[colh];
#pragma unroll
    for (int h = 0; h < 2; ++h)
#pragma unroll
      for (int j = 0; j < 8; ++j) {
        const int n   = h * 16 + rbase + j;
        const int row = row0 + n;
        if (row < nrows) {
          g_float* orow = gfw(out) + (size_t)row * 512;
          orow[colh] = accS[h][j] * inv + b0v;                   // ps
#pragma unroll
          for (int i = 0; i < 3; ++i)
            orow[128 + 3 * colh + i] = accV[i][h][j] * inv;      // pv[n,h,i]
        }
      }
  }
}

// ---------------------------------------------------------------------------
extern "C" void kernel_launch(void* const* d_in, const int* in_sizes, int n_in,
                              void* d_out, int out_size, void* d_ws, size_t ws_size,
                              hipStream_t stream) {
  (void)n_in; (void)out_size; (void)ws_size;
  const float* x      = (const float*)d_in[0];
  const float* fcl1_w = (const float*)d_in[1];
  const float* fcl1_b = (const float*)d_in[2];
  const float* fcl2_w = (const float*)d_in[3];
  const float* fcl2_b = (const float*)d_in[4];
  const float* fcr1_w = (const float*)d_in[5];
  const float* fcr1_b = (const float*)d_in[6];
  const float* fcr2_w = (const float*)d_in[7];
  const float* fcr2_b = (const float*)d_in[8];
  const float* fcp1_w = (const float*)d_in[9];
  const float* fcp1_b = (const float*)d_in[10];
  const float* fcp2_w = (const float*)d_in[11];
  const float* fcp2_b = (const float*)d_in[12];
  const float* Wl0    = (const float*)d_in[13];
  const float* bl0    = (const float*)d_in[14];
  const float* Wl1    = (const float*)d_in[15];
  const float* Wr0    = (const float*)d_in[16];
  const float* br0    = (const float*)d_in[17];
  const float* Wr1    = (const float*)d_in[18];
  const float* Wp0    = (const float*)d_in[19];
  const float* bp0    = (const float*)d_in[20];
  const float* Wp1    = (const float*)d_in[21];
  const float* w_ss   = (const float*)d_in[22];
  const float* w_sv   = (const float*)d_in[23];
  const float* w_vs   = (const float*)d_in[24];
  const float* w_vv   = (const float*)d_in[25];

  // workspace: transposed bf16 weights (B-operand friendly: Wt[n][k])
  bf16_t* p = (bf16_t*)d_ws;
  bf16_t* fcl1_t = p; p += 512 * 512;
  bf16_t* fcl2_t = p; p += 512 * 512;
  bf16_t* fcr1_t = p; p += 512 * 512;
  bf16_t* fcr2_t = p; p += 512 * 512;
  bf16_t* fcp1_t = p; p += 512 * 512;
  bf16_t* fcp2_t = p; p += 512 * 512;
  bf16_t* Wl0_t  = p; p += 256 * 128;
  bf16_t* Wl1_t  = p; p += 256 * 128;
  bf16_t* Wr0_t  = p; p += 256 * 128;
  bf16_t* Wr1_t  = p; p += 256 * 128;
  bf16_t* Wp0_t  = p; p += 256 * 128;
  bf16_t* Wp1_t  = p; p += 256 * 128;

  const int big = 512 * 512, small = 256 * 128;
  dim3 tb(256);
  dim3 gb((big + 255) / 256), gsml((small + 255) / 256);
  transpose_bf16_kernel<<<gb, tb, 0, stream>>>(fcl1_w, fcl1_t, 512, 512);
  transpose_bf16_kernel<<<gb, tb, 0, stream>>>(fcl2_w, fcl2_t, 512, 512);
  transpose_bf16_kernel<<<gb, tb, 0, stream>>>(fcr1_w, fcr1_t, 512, 512);
  transpose_bf16_kernel<<<gb, tb, 0, stream>>>(fcr2_w, fcr2_t, 512, 512);
  transpose_bf16_kernel<<<gb, tb, 0, stream>>>(fcp1_w, fcp1_t, 512, 512);
  transpose_bf16_kernel<<<gb, tb, 0, stream>>>(fcp2_w, fcp2_t, 512, 512);
  transpose_bf16_kernel<<<gsml, tb, 0, stream>>>(Wl0, Wl0_t, 256, 128);
  transpose_bf16_kernel<<<gsml, tb, 0, stream>>>(Wl1, Wl1_t, 256, 128);
  transpose_bf16_kernel<<<gsml, tb, 0, stream>>>(Wr0, Wr0_t, 256, 128);
  transpose_bf16_kernel<<<gsml, tb, 0, stream>>>(Wr1, Wr1_t, 256, 128);
  transpose_bf16_kernel<<<gsml, tb, 0, stream>>>(Wp0, Wp0_t, 256, 128);
  transpose_bf16_kernel<<<gsml, tb, 0, stream>>>(Wp1, Wp1_t, 256, 128);

  const int nrows = in_sizes[0] / 1024;  // N nodes
  const size_t smem =
      (size_t)(32 * 256 + 3 * 32 * 256 + 32 * 512 + 32 * 512 + 32 * 256 +
               3 * 32 * 256 + 32 * 128 + 3 * 32 * 128) * sizeof(bf16_t);  // 224 KB
  hipFuncSetAttribute((const void*)selflayer_kernel,
                      hipFuncAttributeMaxDynamicSharedMemorySize, (int)smem);
  selflayer_kernel<<<dim3((nrows + TILE_M - 1) / TILE_M), dim3(NTHREADS), smem,
                     stream>>>(
      x, fcl1_b, fcl2_b, fcr1_b, fcr2_b, fcp1_b, fcp2_b, bl0, br0, bp0,
      w_ss, w_sv, w_vs, w_vv,
      fcl1_t, fcl2_t, fcr1_t, fcr2_t, fcp1_t, fcp2_t,
      Wl0_t, Wl1_t, Wr0_t, Wr1_t, Wp0_t, Wp1_t,
      (float*)d_out, nrows);
}
